// LSTM_20736102105743
// MI455X (gfx1250) — matrix operand order repaired and checked by reading, not compiled, not run on hardware
//
#include <hip/hip_runtime.h>
#include <hip/hip_bf16.h>
#include <stdint.h>

#define B_   128
#define S_   256
#define I_   1404
#define H_   128
#define E_   128
#define R_   32768      // B*S rows (time-major: r = s*B + b)
#define G4_  512        // 4*H gate width
#define KP_  1408       // I padded to multiple of 32

typedef unsigned short u16;
typedef unsigned int   u32;
typedef __attribute__((ext_vector_type(16))) __bf16 v16bf;
typedef __attribute__((ext_vector_type(8)))  float  v8f;

// ---------- helpers ----------------------------------------------------------

__device__ __forceinline__ u16 f2bf(float f){
  u32 u = __float_as_uint(f);
  u32 r = u + 0x7FFFu + ((u >> 16) & 1u);   // round-to-nearest-even
  return (u16)(r >> 16);
}

__device__ __forceinline__ v8f v8f_zero(){
  v8f z;
#pragma unroll
  for (int i = 0; i < 8; ++i) z[i] = 0.f;
  return z;
}

union FragU { v16bf v; uint4 q[2]; };

__device__ __forceinline__ v8f wmma_bf16(v16bf a, v16bf b, v8f c){
  // (neg_a, A, neg_b, B, c_mod, C, reuse_a, reuse_b)
  return __builtin_amdgcn_wmma_f32_16x16x32_bf16(false, a, false, b, (short)0, c, false, false);
}

// B fragment (32x16, bf16): weight stored row-major W[n][k] so column n is
// contiguous in k.  lane: n = lane&15 ; k blocks [kb,kb+8) and [kb+16,kb+24), kb = k0 + ((lane&16)?8:0).
__device__ __forceinline__ v16bf load_bfrag(const u16* wrow, int k0, int lane){
  FragU u;
  const uint4* p = (const uint4*)(wrow + k0 + ((lane & 16) ? 8 : 0));
  u.q[0] = p[0];
  u.q[1] = p[2];
  return u.v;
}

// A fragment (16x32, bf16) from a bf16 row: k blocks [kb,kb+8) and [kb+16,kb+24)
__device__ __forceinline__ v16bf load_afrag_bf(const u16* arow, int k0, int lane){
  int kb = k0 + ((lane & 16) ? 8 : 0);
  FragU u;
  u.q[0] = *(const uint4*)(arow + kb);
  u.q[1] = *(const uint4*)(arow + kb + 16);
  return u.v;
}

__device__ __forceinline__ uint4 pack8f(const float* p){
  float4 a = *(const float4*)p;
  float4 b = *(const float4*)(p + 4);
  uint4 q;
  q.x = (u32)f2bf(a.x) | ((u32)f2bf(a.y) << 16);
  q.y = (u32)f2bf(a.z) | ((u32)f2bf(a.w) << 16);
  q.z = (u32)f2bf(b.x) | ((u32)f2bf(b.y) << 16);
  q.w = (u32)f2bf(b.z) | ((u32)f2bf(b.w) << 16);
  return q;
}

__device__ __forceinline__ uint4 pack8f_m(const float* row, int kb, int K){
  float v[8];
#pragma unroll
  for (int j = 0; j < 8; ++j) v[j] = (kb + j < K) ? row[kb + j] : 0.f;
  uint4 q;
  q.x = (u32)f2bf(v[0]) | ((u32)f2bf(v[1]) << 16);
  q.y = (u32)f2bf(v[2]) | ((u32)f2bf(v[3]) << 16);
  q.z = (u32)f2bf(v[4]) | ((u32)f2bf(v[5]) << 16);
  q.w = (u32)f2bf(v[6]) | ((u32)f2bf(v[7]) << 16);
  return q;
}

// A fragment from f32 row, converting to bf16 on the fly.
__device__ __forceinline__ v16bf load_afrag_f32(const float* arow, int k0, int lane){
  int kb = k0 + ((lane & 16) ? 8 : 0);
  FragU u;
  u.q[0] = pack8f(arow + kb);
  u.q[1] = pack8f(arow + kb + 16);
  return u.v;
}

__device__ __forceinline__ v16bf load_afrag_f32_m(const float* arow, int k0, int lane, int K){
  int kb = k0 + ((lane & 16) ? 8 : 0);
  FragU u;
  u.q[0] = pack8f_m(arow, kb, K);
  u.q[1] = pack8f_m(arow, kb + 16, K);
  return u.v;
}

__device__ __forceinline__ float sigm_(float x){ return 1.f / (1.f + __expf(-x)); }
__device__ __forceinline__ float tanh_(float x){ return 2.f / (1.f + __expf(-2.f * x)) - 1.f; }

// ---------- kernel 1: weight prep (f32 -> bf16, concat gate weights) ---------
// regions: WgI [512][1408] | WgH [512][128] | bg[512] | Wout [128][256]
//          Wqkv+ao [4][128][128]
__global__ void k_prep(const float* Wf, const float* Wi, const float* Wo, const float* Wc,
                       const float* bfv, const float* biv, const float* bov, const float* bcv,
                       const float* Wout, const float* Wq, const float* Wk, const float* Wv,
                       const float* Wao,
                       u16* WgI, u16* WgH, float* bg, u16* WoutB, u16* WqkvB){
  int id = blockIdx.x * 256 + threadIdx.x;
  if (id < G4_ * KP_){
    int n = id / KP_, k = id % KP_;
    int sel = n >> 7, rrow = n & 127;
    const float* W = (sel == 0) ? Wf : (sel == 1) ? Wi : (sel == 2) ? Wo : Wc;
    WgI[id] = (k < I_) ? f2bf(W[(size_t)rrow * (I_ + H_) + k]) : (u16)0;
    return;
  }
  id -= G4_ * KP_;
  if (id < G4_ * H_){
    int n = id >> 7, k = id & 127;
    int sel = n >> 7, rrow = n & 127;
    const float* W = (sel == 0) ? Wf : (sel == 1) ? Wi : (sel == 2) ? Wo : Wc;
    WgH[id] = f2bf(W[(size_t)rrow * (I_ + H_) + I_ + k]);
    return;
  }
  id -= G4_ * H_;
  if (id < G4_){
    int sel = id >> 7;
    const float* bsrc = (sel == 0) ? bfv : (sel == 1) ? biv : (sel == 2) ? bov : bcv;
    bg[id] = bsrc[id & 127];
    return;
  }
  id -= G4_;
  if (id < E_ * 2 * H_){ WoutB[id] = f2bf(Wout[id]); return; }
  id -= E_ * 2 * H_;
  if (id < 4 * E_ * E_){
    int sel = id >> 14;
    const float* W = (sel == 0) ? Wq : (sel == 1) ? Wk : (sel == 2) ? Wv : Wao;
    WqkvB[id] = f2bf(W[id & 16383]);
  }
}

// ---------- kernel 2: xproj = x @ WgI.T + bg  (M=32768, N=512, K=1404) -------
// grid (256, 4), 256 threads (8 waves).  Row r is time-major: b=r&127, s=r>>7.
__global__ __launch_bounds__(256) void k_xproj(const float* __restrict__ x,
                                               const u16* __restrict__ WgI,
                                               const float* __restrict__ bg,
                                               float* __restrict__ xp){
  int lane = threadIdx.x & 31, wave = threadIdx.x >> 5;
  int m0 = blockIdx.x * 128 + wave * 16;
  int nb = blockIdx.y * 128;
  int r  = m0 + (lane & 15);
  int b  = r & 127, s = r >> 7;
  const float* arow = x + ((size_t)b * S_ + s) * I_;

  v8f acc[8];
#pragma unroll
  for (int i = 0; i < 8; ++i) acc[i] = v8f_zero();

  for (int kk = 0; kk < KP_ / 32; ++kk){
    int k0 = kk * 32;
    v16bf a = (k0 + 32 <= I_) ? load_afrag_f32(arow, k0, lane)
                              : load_afrag_f32_m(arow, k0, lane, I_);
#pragma unroll
    for (int nt = 0; nt < 8; ++nt){
      int n = nb + nt * 16 + (lane & 15);
      v16bf bf_ = load_bfrag(WgI + (size_t)n * KP_, k0, lane);
      acc[nt] = wmma_bf16(a, bf_, acc[nt]);
    }
  }
  int half = (lane & 16) ? 8 : 0;
#pragma unroll
  for (int nt = 0; nt < 8; ++nt){
    int n = nb + nt * 16 + (lane & 15);
    float bias = bg[n];
#pragma unroll
    for (int v = 0; v < 8; ++v){
      int rr = m0 + v + half;
      xp[(size_t)rr * G4_ + n] = acc[nt][v] + bias;
    }
  }
}

// ---------- kernel 3: bidirectional LSTM scan (1 workgroup per direction) ----
// LDS: WgH bf16 [512][128] (131072 B) + h bf16 [128][128] (32768 B) = 163840 B
// wave w: m0 = (w/4)*16, j0 = (w%4)*32, two column groups (j0, j0+16);
// each group owns its 4 gate tiles -> cell update fully in registers.
__global__ __launch_bounds__(1024, 1) void k_scan(const float* __restrict__ xp,
                                                  const u16* __restrict__ WgH,
                                                  const float* __restrict__ h0b,
                                                  const float* __restrict__ c0b,
                                                  u16* __restrict__ hsF,
                                                  u16* __restrict__ hsB){
  extern __shared__ char smem[];
  u16* sW = (u16*)smem;                  // [512][128]
  u16* sH = (u16*)(smem + 131072);       // [128][128]
  const int dir = blockIdx.x;            // 0 = fwd, 1 = bwd
  const int tid = threadIdx.x;
  const int lane = tid & 31, wave = tid >> 5;
  const int m0 = (wave >> 2) * 16;
  const int j0 = (wave & 3) * 32;
  const int half = (lane & 16) ? 8 : 0;

  // stage recurrent weights into LDS (8192 uint4)
  {
    const uint4* src = (const uint4*)WgH;
    uint4* dst = (uint4*)sW;
#pragma unroll
    for (int i = 0; i < 8; ++i) dst[tid + i * 1024] = src[tid + i * 1024];
  }
  // init h
#pragma unroll
  for (int i = 0; i < 16; ++i){
    int idx = tid + i * 1024;
    sH[idx] = dir ? f2bf(h0b[idx & 127]) : (u16)0;
  }
  // init c in registers: element (g,v) <-> row m0+v+half, col j0+g*16+(lane&15)
  float creg[2][8];
#pragma unroll
  for (int g = 0; g < 2; ++g){
    int jc = j0 + g * 16 + (lane & 15);
    float cinit = dir ? c0b[jc] : 0.f;
#pragma unroll
    for (int v = 0; v < 8; ++v) creg[g][v] = cinit;
  }
  __syncthreads();

  u16* hs = dir ? hsB : hsF;

  for (int step = 0; step < S_; ++step){
    int t = dir ? (S_ - 1 - step) : step;
    const float* xpt = xp + (size_t)t * B_ * G4_;

    v8f acc[2][4];
    // C init = xproj slice (f32 accumulate continues the input projection)
#pragma unroll
    for (int g = 0; g < 2; ++g)
#pragma unroll
      for (int gate = 0; gate < 4; ++gate){
        int n = gate * 128 + j0 + g * 16 + (lane & 15);
        v8f c;
#pragma unroll
        for (int v = 0; v < 8; ++v) c[v] = xpt[(size_t)(m0 + v + half) * G4_ + n];
        acc[g][gate] = c;
      }
    // recurrent GEMM: h(128x128) x WgH^T -> gates(128x512)
#pragma unroll
    for (int kk = 0; kk < 4; ++kk){
      int k0 = kk * 32;
      v16bf a = load_afrag_bf(sH + (size_t)(m0 + (lane & 15)) * H_, k0, lane);
#pragma unroll
      for (int g = 0; g < 2; ++g)
#pragma unroll
        for (int gate = 0; gate < 4; ++gate){
          int n = gate * 128 + j0 + g * 16 + (lane & 15);
          v16bf bf_ = load_bfrag(sW + (size_t)n * H_, k0, lane);
          acc[g][gate] = wmma_bf16(a, bf_, acc[g][gate]);
        }
    }
    __syncthreads();   // everyone done reading old h

    // cell update in registers, write new h (LDS + global hs, bf16)
#pragma unroll
    for (int g = 0; g < 2; ++g){
      int jc = j0 + g * 16 + (lane & 15);
#pragma unroll
      for (int v = 0; v < 8; ++v){
        int rrow = m0 + v + half;
        float f  = sigm_(acc[g][0][v]);
        float in = sigm_(acc[g][1][v]);
        float o  = sigm_(acc[g][2][v]);
        float ch = tanh_(acc[g][3][v]);
        float c  = f * creg[g][v] + in * ch;
        creg[g][v] = c;
        u16 hb = f2bf(o * tanh_(c));
        sH[(size_t)rrow * H_ + jc] = hb;
        hs[((size_t)t * B_ + rrow) * H_ + jc] = hb;
      }
    }
    __syncthreads();
  }
}

// ---------- kernel 4: y = [fwd|bwd] @ Wout.T + bout (M=32768,N=128,K=256) ----
__global__ __launch_bounds__(256) void k_ygemm(const u16* __restrict__ hsF,
                                               const u16* __restrict__ hsB,
                                               const u16* __restrict__ WoutB,
                                               const float* __restrict__ bout,
                                               float* __restrict__ y){
  int lane = threadIdx.x & 31, wave = threadIdx.x >> 5;
  int m0 = blockIdx.x * 128 + wave * 16;
  int r  = m0 + (lane & 15);
  const u16* aF = hsF + (size_t)r * H_;
  const u16* aB = hsB + (size_t)r * H_;

  v8f acc[8];
#pragma unroll
  for (int i = 0; i < 8; ++i) acc[i] = v8f_zero();

#pragma unroll
  for (int kk = 0; kk < 8; ++kk){
    int k0 = kk * 32;
    v16bf a = (k0 < 128) ? load_afrag_bf(aF, k0, lane) : load_afrag_bf(aB, k0 - 128, lane);
#pragma unroll
    for (int nt = 0; nt < 8; ++nt){
      int n = nt * 16 + (lane & 15);
      v16bf bf_ = load_bfrag(WoutB + (size_t)n * 256, k0, lane);
      acc[nt] = wmma_bf16(a, bf_, acc[nt]);
    }
  }
  int half = (lane & 16) ? 8 : 0;
#pragma unroll
  for (int nt = 0; nt < 8; ++nt){
    int n = nt * 16 + (lane & 15);
    float bias = bout[n];
#pragma unroll
    for (int v = 0; v < 8; ++v)
      y[(size_t)(m0 + v + half) * E_ + n] = acc[nt][v] + bias;
  }
}

// ---------- kernel 5/6: LayerNorm stats (deterministic two-stage) ------------
__global__ __launch_bounds__(256) void k_stats1(const float* __restrict__ y,
                                                float* __restrict__ psum,
                                                float* __restrict__ psq){
  __shared__ float ls[256], lq[256];
  int e = threadIdx.x & 127, sub = threadIdx.x >> 7;
  float s = 0.f, q = 0.f;
  for (int i = 0; i < 128; ++i){
    int r = blockIdx.x * 256 + sub * 128 + i;
    float v = y[(size_t)r * E_ + e];
    s += v; q += v * v;
  }
  ls[threadIdx.x] = s; lq[threadIdx.x] = q;
  __syncthreads();
  if (threadIdx.x < 128){
    psum[blockIdx.x * 128 + e] = ls[threadIdx.x] + ls[threadIdx.x + 128];
    psq [blockIdx.x * 128 + e] = lq[threadIdx.x] + lq[threadIdx.x + 128];
  }
}

__global__ void k_stats2(const float* __restrict__ psum, const float* __restrict__ psq,
                         float* __restrict__ sums){
  int e = threadIdx.x;
  float s = 0.f, q = 0.f;
  for (int i = 0; i < 128; ++i){ s += psum[i * 128 + e]; q += psq[i * 128 + e]; }
  sums[e] = s; sums[128 + e] = q;
}

__global__ __launch_bounds__(256) void k_norm(float* __restrict__ y,
                                              const float* __restrict__ sums,
                                              const float* __restrict__ gamma,
                                              const float* __restrict__ beta){
  size_t idx = (size_t)blockIdx.x * 256 + threadIdx.x;
  int e = (int)(idx & 127);
  float mean = sums[e] * (1.f / 32768.f);
  float var  = sums[128 + e] * (1.f / 32768.f) - mean * mean;
  float is   = rsqrtf(var + 1e-5f);
  y[idx] = (y[idx] - mean) * is * gamma[e] + beta[e];
}

// ---------- kernel 7: generic GEMM C = A(f32) @ W(bf16,row-major).T + bias ---
// M=32768, N=128, K=128.  Used for Q/K/V and the attn output projection.
__global__ __launch_bounds__(256) void k_gemm128(const float* __restrict__ A,
                                                 const u16* __restrict__ W,
                                                 const float* __restrict__ bias,
                                                 float* __restrict__ C){
  int lane = threadIdx.x & 31, wave = threadIdx.x >> 5;
  int m0 = blockIdx.x * 128 + wave * 16;
  const float* arow = A + (size_t)(m0 + (lane & 15)) * 128;

  v8f acc[8];
#pragma unroll
  for (int i = 0; i < 8; ++i) acc[i] = v8f_zero();

#pragma unroll
  for (int kk = 0; kk < 4; ++kk){
    int k0 = kk * 32;
    v16bf a = load_afrag_f32(arow, k0, lane);
#pragma unroll
    for (int nt = 0; nt < 8; ++nt){
      int n = nt * 16 + (lane & 15);
      v16bf bf_ = load_bfrag(W + (size_t)n * 128, k0, lane);
      acc[nt] = wmma_bf16(a, bf_, acc[nt]);
    }
  }
  int half = (lane & 16) ? 8 : 0;
#pragma unroll
  for (int nt = 0; nt < 8; ++nt){
    int n = nt * 16 + (lane & 15);
    float b = bias[n];
#pragma unroll
    for (int v = 0; v < 8; ++v)
      C[(size_t)(m0 + v + half) * 128 + n] = acc[nt][v] + b;
  }
}

// ---------- kernel 8: per-token 8-head cross-head attention ------------------
__global__ __launch_bounds__(128) void k_attn(const float* __restrict__ Q,
                                              const float* __restrict__ K,
                                              const float* __restrict__ V,
                                              float* __restrict__ attn){
  __shared__ float sq[128], sk[128], sv[128], sc[64], sp[64];
  size_t r = blockIdx.x;
  int t = threadIdx.x;
  sq[t] = Q[r * 128 + t];
  sk[t] = K[r * 128 + t];
  sv[t] = V[r * 128 + t];
  __syncthreads();
  if (t < 64){
    int h = t >> 3, e = t & 7;
    float d = 0.f;
#pragma unroll
    for (int dd = 0; dd < 16; ++dd) d += sq[h * 16 + dd] * sk[e * 16 + dd];
    sc[t] = d * 0.25f;                       // / sqrt(16)
  }
  __syncthreads();
  if (t < 8){
    float m = -1e30f;
#pragma unroll
    for (int e = 0; e < 8; ++e) m = fmaxf(m, sc[t * 8 + e]);
    float sum = 0.f;
#pragma unroll
    for (int e = 0; e < 8; ++e){ float p = __expf(sc[t * 8 + e] - m); sp[t * 8 + e] = p; sum += p; }
    float inv = 1.f / sum;
#pragma unroll
    for (int e = 0; e < 8; ++e) sp[t * 8 + e] *= inv;
  }
  __syncthreads();
  int h = t >> 4, d = t & 15;
  float a = 0.f;
#pragma unroll
  for (int e = 0; e < 8; ++e) a += sp[h * 8 + e] * sv[e * 16 + d];
  attn[r * 128 + t] = a;
}

// ---------- kernel 9: row max + (s,b) -> (b,s) remap -------------------------
__global__ __launch_bounds__(256) void k_rowmax(const float* __restrict__ z,
                                                float* __restrict__ out){
  int r = blockIdx.x * 256 + threadIdx.x;
  const float* row = z + (size_t)r * 128;
  float m = -1e30f;
#pragma unroll 8
  for (int n = 0; n < 128; ++n) m = fmaxf(m, row[n]);
  int b = r & 127, s = r >> 7;
  out[(size_t)b * S_ + s] = m;
}

// ---------- workspace layout (bytes) -----------------------------------------
#define OFF_WGI    0u            // 512*1408*2  = 1441792
#define OFF_WGH    1441792u      // 512*128*2   =  131072
#define OFF_BG     1572864u      // 512*4       =    2048
#define OFF_WOUT   1574912u      // 128*256*2   =   65536
#define OFF_WQKV   1640448u      // 4*128*128*2 =  131072
#define OFF_STATS  1771520u      // psum 65536 | psq 65536 | sums 1024 (pad->139264)
#define OFF_BIG    1910784u      // xproj 64MB ; later Q/K/V (3*16MB)
#define OFF_HSF    69019648u     // hsF bf16 16MB ; later z (f32 16MB)
#define OFF_HSB    85796864u     // hsB bf16 16MB
#define OFF_Y      102574080u    // y f32 16MB ; later attn

extern "C" void kernel_launch(void* const* d_in, const int* in_sizes, int n_in,
                              void* d_out, int out_size, void* d_ws, size_t ws_size,
                              hipStream_t stream){
  const float* x    = (const float*)d_in[0];
  const float* Wf   = (const float*)d_in[1];
  const float* bfv  = (const float*)d_in[2];
  const float* Wi   = (const float*)d_in[3];
  const float* biv  = (const float*)d_in[4];
  const float* Wo   = (const float*)d_in[5];
  const float* bov  = (const float*)d_in[6];
  const float* Wc   = (const float*)d_in[7];
  const float* bcv  = (const float*)d_in[8];
  const float* Wout = (const float*)d_in[9];
  const float* bout = (const float*)d_in[10];
  const float* h0b  = (const float*)d_in[11];
  const float* c0b  = (const float*)d_in[12];
  const float* gamma= (const float*)d_in[13];
  const float* beta = (const float*)d_in[14];
  const float* Wq   = (const float*)d_in[15];
  const float* bq   = (const float*)d_in[16];
  const float* Wk   = (const float*)d_in[17];
  const float* bk   = (const float*)d_in[18];
  const float* Wv   = (const float*)d_in[19];
  const float* bv   = (const float*)d_in[20];
  const float* Wao  = (const float*)d_in[21];
  const float* bao  = (const float*)d_in[22];
  float* out = (float*)d_out;
  char*  ws  = (char*)d_ws;

  u16*   WgI   = (u16*)  (ws + OFF_WGI);
  u16*   WgH   = (u16*)  (ws + OFF_WGH);
  float* bg    = (float*)(ws + OFF_BG);
  u16*   WoutB = (u16*)  (ws + OFF_WOUT);
  u16*   WqkvB = (u16*)  (ws + OFF_WQKV);
  float* psum  = (float*)(ws + OFF_STATS);
  float* psq   = (float*)(ws + OFF_STATS + 65536);
  float* sums  = (float*)(ws + OFF_STATS + 131072);
  float* xproj = (float*)(ws + OFF_BIG);
  u16*   hsF   = (u16*)  (ws + OFF_HSF);
  u16*   hsB   = (u16*)  (ws + OFF_HSB);
  float* y     = (float*)(ws + OFF_Y);
  float* Qm    = (float*)(ws + OFF_BIG);                 // reuse xproj region
  float* Km    = (float*)(ws + OFF_BIG + 16777216u);
  float* Vm    = (float*)(ws + OFF_BIG + 33554432u);
  float* attn  = y;                                      // reuse y region
  float* z     = (float*)(ws + OFF_HSF);                 // reuse hs region

  // 1. weight conversion / packing
  k_prep<<<3458, 256, 0, stream>>>(Wf, Wi, Wo, Wc, bfv, biv, bov, bcv,
                                   Wout, Wq, Wk, Wv, Wao,
                                   WgI, WgH, bg, WoutB, WqkvB);
  // 2. input projection of gates (shared by both directions)
  k_xproj<<<dim3(256, 4), 256, 0, stream>>>(x, WgI, bg, xproj);
  // 3. bidirectional LSTM scan (block 0 = fwd, block 1 = bwd)
  k_scan<<<2, 1024, 163840, stream>>>(xproj, WgH, h0b, c0b, hsF, hsB);
  // 4. output projection
  k_ygemm<<<256, 256, 0, stream>>>(hsF, hsB, WoutB, bout, y);
  // 5/6. layernorm (deterministic reduction), normalize in place
  k_stats1<<<128, 256, 0, stream>>>(y, psum, psq);
  k_stats2<<<1, 128, 0, stream>>>(psum, psq, sums);
  k_norm<<<16384, 256, 0, stream>>>(y, sums, gamma, beta);
  // 7. Q/K/V projections
  k_gemm128<<<256, 256, 0, stream>>>(y, WqkvB +      0, bq, Qm);
  k_gemm128<<<256, 256, 0, stream>>>(y, WqkvB + 16384, bk, Km);
  k_gemm128<<<256, 256, 0, stream>>>(y, WqkvB + 32768, bv, Vm);
  // 8. per-token cross-head attention
  k_attn<<<32768, 128, 0, stream>>>(Qm, Km, Vm, attn);
  // 9. output projection + row max
  k_gemm128<<<256, 256, 0, stream>>>(attn, WqkvB + 49152, bao, z);
  k_rowmax<<<128, 256, 0, stream>>>(z, out);
}
